// ClassifierB_14267881357429
// MI455X (gfx1250) — matrix operand order repaired and checked by reading, not compiled, not run on hardware
//
#include <hip/hip_runtime.h>

// ---------------------------------------------------------------------------
// ST-GCN classifier forward for MI455X (gfx1250), token-major [M=1536, E=2048]
// Big GEMMs: v_wmma_f32_16x16x32_bf16, fp32 accumulate.
// Tile staging: Tensor Data Mover (tensor_load_to_lds) from pre-converted
// bf16 weights/activations, with LDS row padding done by the TDM descriptor.
// Fallback path (small ws / no builtin): fp32->bf16 VALU staging.
// ---------------------------------------------------------------------------

typedef __attribute__((ext_vector_type(16))) __bf16 v16bf;
typedef __attribute__((ext_vector_type(8)))  float  v8f;
typedef unsigned int uint32x4 __attribute__((ext_vector_type(4)));
typedef int          int32x4  __attribute__((ext_vector_type(4)));
typedef int          int32x8  __attribute__((ext_vector_type(8)));

#define E_DIM 2048
#define M_DIM 1536   // B*T*P = 4*32*12
#define T_DIM 32
#define P_DIM 12
#define BM 128
#define BN 128
#define BK 32
#define LDS_STRIDE 40   // bf16 elems per LDS row; 80B, multiple of 16B

#if __has_builtin(__builtin_amdgcn_tensor_load_to_lds) && __has_builtin(__builtin_amdgcn_s_wait_tensorcnt)
#define HAVE_TDM 1
#else
#define HAVE_TDM 0
#endif

__device__ __forceinline__ unsigned short f2bf(float x) {
  unsigned u = __float_as_uint(x);
  u += 0x7FFFu + ((u >> 16) & 1u);          // round-to-nearest-even
  return (unsigned short)(u >> 16);
}

struct __align__(16) U128 { unsigned int a, b, c, d; };
union FragAB { v16bf v; U128 q[2]; };

// ---------------------------------------------------------------------------
// TDM: load one 128-row x 32-col bf16 tile (row stride 2048 elems in memory)
// into LDS at lds_addr with +16B padding after each 64B row (stride 40 elems).
// D# layout per CDNA5 ISA ch.8 (async_tensor): group0 = count/lds/global/type,
// group1 = data_size|pad | tensor_dim0/1 | tile_dim0/1/2 | dim0_stride.
// ---------------------------------------------------------------------------
__device__ __forceinline__ void tdm_load_tile(unsigned lds_addr, const void* gptr)
{
#if HAVE_TDM
  unsigned long long ga = (unsigned long long)(size_t)gptr;
  uint32x4 g0;
  g0[0] = 1u;                                          // count=1, user descriptor
  g0[1] = lds_addr;                                    // LDS byte address
  g0[2] = (unsigned)(ga & 0xFFFFFFFFu);                // global_addr[31:0]
  g0[3] = (unsigned)((ga >> 32) & 0x01FFFFFFu) | (2u << 30);  // addr[56:32] | type=2
  int32x8 g1;
  // data_size=1 (2B) | pad_enable | pad_interval=3 (16 DW = 64B) | pad_amount=3 (4 DW = 16B)
  g1[0] = (int)((1u << 16) | (1u << 20) | (3u << 22) | (3u << 25));
  g1[1] = (int)(((unsigned)E_DIM & 0xFFFFu) << 16);    // tensor_dim0[15:0] @ bits[63:48]
  g1[2] = (int)(((unsigned)E_DIM & 0xFFFFu) << 16);    // dim0[31:16]=0 | tensor_dim1[15:0]
  g1[3] = (int)((unsigned)BK << 16);                   // dim1[31:16]=0 | tile_dim0=32
  g1[4] = (int)BM;                                     // tile_dim1=128 | tile_dim2=0
  g1[5] = (int)E_DIM;                                  // tensor_dim0_stride[31:0]
  g1[6] = 0;                                           // stride hi | dim1_stride lo
  g1[7] = 0;
  int32x4 z4 = {0, 0, 0, 0};
#if __clang_major__ >= 23
  int32x8 z8 = {0, 0, 0, 0, 0, 0, 0, 0};
  __builtin_amdgcn_tensor_load_to_lds(g0, g1, z4, z4, z8, 0);
#else
  __builtin_amdgcn_tensor_load_to_lds(g0, g1, z4, z4, 0);
#endif
#else
  (void)lds_addr; (void)gptr;
#endif
}

// MODE 0: embed   epilogue: relu(acc + bias)                -> Out (+Obf)
// MODE 1: sconv   epilogue: acc + bias                      -> Out
// MODE 2: tconv   3 t-shifted A passes, epilogue:
//                 relu(bn2(acc + bias)) + residual, in place -> Out (+Obf)
template<int MODE, bool TDM>
__global__ __launch_bounds__(256) void gemm_bf16_wmma(
    const float* __restrict__ A, const float* __restrict__ Bm,          // fp32 (fallback)
    const unsigned short* __restrict__ Abf,
    const unsigned short* __restrict__ Bbf,                             // bf16 (TDM)
    const float* __restrict__ bias, const float* __restrict__ bnG,
    const float* __restrict__ bnB, float* __restrict__ Out,
    unsigned short* __restrict__ Obf)
{
  __shared__ __align__(16) unsigned short sA[BM * LDS_STRIDE];
  __shared__ __align__(16) unsigned short sB[BN * LDS_STRIDE];

  const int tid  = threadIdx.x;
  const int lane = tid & 31, wid = tid >> 5;
  const int half = lane >> 4, l16 = lane & 15;
  const int wm = wid >> 2, wn = wid & 3;         // 2 x 4 wave grid
  const int mBase = blockIdx.y * BM;
  const int nBase = blockIdx.x * BN;

  const v8f vz = {0.f, 0.f, 0.f, 0.f, 0.f, 0.f, 0.f, 0.f};
  v8f acc[4][2];
#pragma unroll
  for (int i = 0; i < 4; ++i) { acc[i][0] = vz; acc[i][1] = vz; }

  const int KT = (MODE == 2) ? 3 : 1;
  for (int kk = 0; kk < KT; ++kk) {
    const int dt = kk - 1;
    for (int k0 = 0; k0 < E_DIM; k0 += BK) {
      __syncthreads();   // all waves done reading LDS from previous iteration
      if (TDM) {
        if (wid == 0) {
          // ---- A tile ----
          const unsigned short* aptr;
          if (MODE == 2)
            aptr = Abf + ((long long)(mBase + dt * P_DIM) * E_DIM + k0);
          else
            aptr = Abf + ((size_t)mBase * E_DIM + k0);
          tdm_load_tile((unsigned)(size_t)sA, aptr);
          // ---- B tile (all modes pre-laid-out as [n][k] bf16) ----
          const unsigned short* bptr;
          if (MODE == 2)
            bptr = Bbf + (((size_t)kk * E_DIM + nBase) * E_DIM + k0);
          else
            bptr = Bbf + ((size_t)nBase * E_DIM + k0);
          tdm_load_tile((unsigned)(size_t)sB, bptr);
#if HAVE_TDM
          __builtin_amdgcn_s_wait_tensorcnt(0);
#endif
          if (MODE == 2) {
            // zero the <=12 rows whose shifted t falls outside the batch
            int bl = mBase % (T_DIM * P_DIM);              // tiles are batch-aligned
            bool firstT = (bl == 0)      && (dt < 0);
            bool lastT  = (bl == 2 * BM) && (dt > 0);
            if ((firstT || lastT) && lane < P_DIM) {
              int r = firstT ? lane : (BM - P_DIM + lane);
              U128 z = {0u, 0u, 0u, 0u};
              *(U128*)&sA[r * LDS_STRIDE +  0] = z;
              *(U128*)&sA[r * LDS_STRIDE +  8] = z;
              *(U128*)&sA[r * LDS_STRIDE + 16] = z;
              *(U128*)&sA[r * LDS_STRIDE + 24] = z;
            }
          }
        }
      } else {
        // ---- fallback: stage fp32 -> bf16 with VALU ----
        for (int idx = tid; idx < BM * BK; idx += 256) {
          int r = idx >> 5, c = idx & 31;
          int grow = mBase + r, gk = k0 + c;
          float v;
          if (MODE == 2) {
            int t  = (grow / P_DIM) % T_DIM;
            int tt = t + dt;
            v = (tt >= 0 && tt < T_DIM)
                  ? A[(size_t)(grow + dt * P_DIM) * E_DIM + gk] : 0.f;
          } else {
            v = A[(size_t)grow * E_DIM + gk];
          }
          sA[r * LDS_STRIDE + c] = f2bf(v);
        }
        for (int idx = tid; idx < BN * BK; idx += 256) {
          int n = idx >> 5, c = idx & 31;
          int gn = nBase + n, gk = k0 + c;
          float v;
          if (MODE == 0)      v = Bm[(size_t)gk * E_DIM + gn];
          else if (MODE == 1) v = Bm[(size_t)gn * E_DIM + gk];
          else                v = Bm[((size_t)gn * E_DIM + gk) * 3 + kk];
          sB[n * LDS_STRIDE + c] = f2bf(v);
        }
      }
      __syncthreads();

      // ---- fragments per ISA VGPR layouts ----
      FragAB af[4], bfr[2];
#pragma unroll
      for (int i = 0; i < 4; ++i) {
        int arow = wm * 64 + i * 16 + l16;
        const unsigned short* base = &sA[arow * LDS_STRIDE + half * 8];
        af[i].q[0] = *(const U128*)base;          // K = 8*half .. +7
        af[i].q[1] = *(const U128*)(base + 16);   // K = 16 + 8*half .. +7
      }
#pragma unroll
      for (int j = 0; j < 2; ++j) {
        int bcol = wn * 32 + j * 16 + l16;
        const unsigned short* base = &sB[bcol * LDS_STRIDE + half * 8];
        bfr[j].q[0] = *(const U128*)base;         // K = 8*half .. +7
        bfr[j].q[1] = *(const U128*)(base + 16);  // K = 16 + 8*half .. +7
      }
#pragma unroll
      for (int i = 0; i < 4; ++i)
#pragma unroll
        for (int j = 0; j < 2; ++j)
          acc[i][j] = __builtin_amdgcn_wmma_f32_16x16x32_bf16(
              false, af[i].v, false, bfr[j].v, (short)0, acc[i][j], false, false);
    }
  }

  // ---- epilogue ----
#pragma unroll
  for (int i = 0; i < 4; ++i) {
#pragma unroll
    for (int j = 0; j < 2; ++j) {
      int col = nBase + wn * 32 + j * 16 + l16;
#pragma unroll
      for (int r = 0; r < 8; ++r) {
        int row = mBase + wm * 64 + i * 16 + half * 8 + r;
        size_t oidx = (size_t)row * E_DIM + col;
        float v = acc[i][j][r];
        float y;
        if (MODE == 0) {
          y = fmaxf(v + bias[col], 0.f);
          Out[oidx] = y;
        } else if (MODE == 1) {
          y = v + bias[col];
          Out[oidx] = y;
        } else {
          float sc = bnG[col] * rsqrtf(1.f + 1e-5f);
          y = fmaxf((v + bias[col]) * sc + bnB[col], 0.f) + Out[oidx];
          Out[oidx] = y;           // residual, in place
        }
        if (TDM && MODE != 1) Obf[oidx] = f2bf(y);   // bf16 mirror for next GEMM's A
      }
    }
  }
}

// ---------------- conversion pre-passes (TDM path only) ----------------
__global__ __launch_bounds__(256) void cvt_bf16_kernel(
    const float* __restrict__ in, unsigned short* __restrict__ out, int n)
{
  int i = blockIdx.x * 256 + threadIdx.x;
  if (i < n) out[i] = f2bf(in[i]);
}

// emb_w [F,E] -> [E][F] bf16
__global__ __launch_bounds__(256) void cvt_transpose_kernel(
    const float* __restrict__ in, unsigned short* __restrict__ out)
{
  int i = blockIdx.x * 256 + threadIdx.x;      // i = e*F + f
  int e = i >> 11, f = i & (E_DIM - 1);
  out[i] = f2bf(in[(size_t)f * E_DIM + e]);
}

// tconv_w [O][I][3] -> [kk][O][I] bf16
__global__ __launch_bounds__(256) void cvt_tconv_kernel(
    const float* __restrict__ in, unsigned short* __restrict__ out)
{
  size_t i = (size_t)blockIdx.x * 256 + threadIdx.x;   // i = (kk*E + o)*E + ii
  size_t EF = (size_t)E_DIM * E_DIM;
  int kk = (int)(i / EF);
  size_t rem = i - (size_t)kk * EF;
  out[i] = f2bf(in[rem * 3 + kk]);
}

// ---------------- small glue kernels ----------------
__global__ __launch_bounds__(256) void thetaphi_kernel(
    const float* __restrict__ X, const float* __restrict__ tw,
    const float* __restrict__ tb, const float* __restrict__ pw,
    const float* __restrict__ pb, float* __restrict__ theta,
    float* __restrict__ phi)
{
  __shared__ float red[256];
  int m = blockIdx.x, tid = threadIdx.x;
  float st = 0.f, sp = 0.f;
  for (int e = tid; e < E_DIM; e += 256) {
    float v = X[(size_t)m * E_DIM + e];
    st += v * tw[e];
    sp += v * pw[e];
  }
  red[tid] = st; __syncthreads();
  for (int s = 128; s > 0; s >>= 1) { if (tid < s) red[tid] += red[tid + s]; __syncthreads(); }
  if (tid == 0) theta[m] = red[0] + tb[0];
  __syncthreads();
  red[tid] = sp; __syncthreads();
  for (int s = 128; s > 0; s >>= 1) { if (tid < s) red[tid] += red[tid + s]; __syncthreads(); }
  if (tid == 0) phi[m] = red[0] + pb[0];
}

__global__ __launch_bounds__(256) void graph_softmax_kernel(
    const float* __restrict__ theta, const float* __restrict__ phi,
    float* __restrict__ C)
{
  __shared__ float f[P_DIM][P_DIM];
  int b = blockIdx.x, tid = threadIdx.x;
  if (tid < P_DIM * P_DIM) {
    int p = tid / P_DIM, q = tid % P_DIM;
    float acc = 0.f;
    for (int t = 0; t < T_DIM; ++t)
      acc += theta[(b * T_DIM + t) * P_DIM + p] * phi[(b * T_DIM + t) * P_DIM + q];
    f[p][q] = acc;
  }
  __syncthreads();
  if (tid < P_DIM) {
    int q = tid;
    float mx = -3.4e38f;
    for (int p = 0; p < P_DIM; ++p) mx = fmaxf(mx, f[p][q]);
    float s = 0.f;
    for (int p = 0; p < P_DIM; ++p) s += __expf(f[p][q] - mx);
    float inv = 1.f / s;
    for (int p = 0; p < P_DIM; ++p)
      C[b * 144 + p * P_DIM + q] = __expf(f[p][q] - mx) * inv;
  }
}

__global__ __launch_bounds__(256) void mix_bn_relu_kernel(
    const float* __restrict__ S, const float* __restrict__ C,
    const float* __restrict__ g, const float* __restrict__ bb,
    float* __restrict__ H, unsigned short* __restrict__ Hbf)
{
  size_t idx = (size_t)blockIdx.x * 256 + threadIdx.x;
  int o  = (int)(idx & (E_DIM - 1));
  int m  = (int)(idx >> 11);
  int q  = m % P_DIM;
  int bt = m / P_DIM;
  int b  = bt / T_DIM;
  const float* Cb = C + b * 144;
  int base = bt * P_DIM;
  float acc = 0.f;
#pragma unroll
  for (int p = 0; p < P_DIM; ++p)
    acc += S[(size_t)(base + p) * E_DIM + o] * Cb[p * P_DIM + q];
  float sc = g[o] * rsqrtf(1.f + 1e-5f);
  float y = fmaxf(acc * sc + bb[o], 0.f);
  H[idx] = y;
  if (Hbf) Hbf[idx] = f2bf(y);
}

__global__ __launch_bounds__(256) void action_head_kernel(
    const float* __restrict__ X, const float* __restrict__ w,
    const float* __restrict__ b, float* __restrict__ out)
{
  __shared__ float red[256];
  int m = blockIdx.x, tid = threadIdx.x;
  float a[9];
#pragma unroll
  for (int j = 0; j < 9; ++j) a[j] = 0.f;
  for (int e = tid; e < E_DIM; e += 256) {
    float v = X[(size_t)m * E_DIM + e];
#pragma unroll
    for (int j = 0; j < 9; ++j) a[j] += v * w[e * 9 + j];
  }
  for (int j = 0; j < 9; ++j) {
    red[tid] = a[j]; __syncthreads();
    for (int s = 128; s > 0; s >>= 1) { if (tid < s) red[tid] += red[tid + s]; __syncthreads(); }
    if (tid == 0) out[m * 9 + j] = red[0] + b[j];
    __syncthreads();
  }
}

__global__ __launch_bounds__(256) void activity_head_kernel(
    const float* __restrict__ X, const float* __restrict__ w,
    const float* __restrict__ b, float* __restrict__ out)
{
  __shared__ float red[256];
  int bt = blockIdx.x, tid = threadIdx.x;
  float a[8];
#pragma unroll
  for (int j = 0; j < 8; ++j) a[j] = 0.f;
  for (int e = tid; e < E_DIM; e += 256) {
    float mx = X[(size_t)(bt * P_DIM + 0) * E_DIM + e];
#pragma unroll
    for (int p = 1; p < P_DIM; ++p)
      mx = fmaxf(mx, X[(size_t)(bt * P_DIM + p) * E_DIM + e]);
#pragma unroll
    for (int j = 0; j < 8; ++j) a[j] += mx * w[e * 8 + j];
  }
  for (int j = 0; j < 8; ++j) {
    red[tid] = a[j]; __syncthreads();
    for (int s = 128; s > 0; s >>= 1) { if (tid < s) red[tid] += red[tid + s]; __syncthreads(); }
    if (tid == 0) out[bt * 8 + j] = red[0] + b[j];
    __syncthreads();
  }
}

extern "C" void kernel_launch(void* const* d_in, const int* in_sizes, int n_in,
                              void* d_out, int out_size, void* d_ws, size_t ws_size,
                              hipStream_t stream)
{
  (void)in_sizes; (void)n_in; (void)out_size;

  const float* feature = (const float*)d_in[0];
  const float* emb_w   = (const float*)d_in[1];
  const float* emb_b   = (const float*)d_in[2];
  const float* act_w   = (const float*)d_in[27];
  const float* act_b   = (const float*)d_in[28];
  const float* acty_w  = (const float*)d_in[29];
  const float* acty_b  = (const float*)d_in[30];

  const size_t MF = (size_t)M_DIM * E_DIM;     // 3,145,728
  const size_t EF = (size_t)E_DIM * E_DIM;     // 4,194,304

  // fp32 region
  float* X  = (float*)d_ws;
  float* S  = X + MF;
  float* H  = S + MF;
  float* th = H + MF;
  float* ph = th + M_DIM;
  float* Cg = ph + M_DIM;
  float* fp32_end = Cg + 1024;

  // bf16 region (TDM path)
  unsigned short* Fbf  = (unsigned short*)fp32_end;
  unsigned short* Xbf  = Fbf + MF;
  unsigned short* Hbf  = Xbf + MF;
  unsigned short* embT = Hbf + MF + 32768;     // 64KB tail pad for shifted reads
  unsigned short* scW[2] = { embT + EF, embT + EF + EF };
  unsigned short* tcW[2] = { scW[1] + EF, scW[1] + EF + 3 * EF };
  size_t need = (size_t)((unsigned short*)(tcW[1] + 3 * EF) - (unsigned short*)d_ws) * 2;

  const bool use_tdm = HAVE_TDM && (ws_size >= need);

  dim3 ggrid(E_DIM / BN, M_DIM / BM);          // (16, 12)

  if (use_tdm) {
    cvt_bf16_kernel<<<(int)(MF / 256), 256, 0, stream>>>(feature, Fbf, (int)MF);
    cvt_transpose_kernel<<<(int)(EF / 256), 256, 0, stream>>>(emb_w, embT);
    for (int blk = 0; blk < 2; ++blk) {
      const float* sconv_w = (const float*)d_in[3 + blk * 12 + 4];
      const float* tconv_w = (const float*)d_in[3 + blk * 12 + 8];
      cvt_bf16_kernel<<<(int)(EF / 256), 256, 0, stream>>>(sconv_w, scW[blk], (int)EF);
      cvt_tconv_kernel<<<(int)(3 * EF / 256), 256, 0, stream>>>(tconv_w, tcW[blk]);
    }
    gemm_bf16_wmma<0, true><<<ggrid, 256, 0, stream>>>(
        nullptr, nullptr, Fbf, embT, emb_b, nullptr, nullptr, X, Xbf);
  } else {
    gemm_bf16_wmma<0, false><<<ggrid, 256, 0, stream>>>(
        feature, emb_w, nullptr, nullptr, emb_b, nullptr, nullptr, X, nullptr);
  }

  for (int blk = 0; blk < 2; ++blk) {
    int o = 3 + blk * 12;   // _block_params insertion order
    const float* theta_w = (const float*)d_in[o + 0];
    const float* theta_b = (const float*)d_in[o + 1];
    const float* phi_w   = (const float*)d_in[o + 2];
    const float* phi_b   = (const float*)d_in[o + 3];
    const float* sconv_w = (const float*)d_in[o + 4];
    const float* sconv_b = (const float*)d_in[o + 5];
    const float* bn1_g   = (const float*)d_in[o + 6];
    const float* bn1_b   = (const float*)d_in[o + 7];
    const float* tconv_w = (const float*)d_in[o + 8];
    const float* tconv_b = (const float*)d_in[o + 9];
    const float* bn2_g   = (const float*)d_in[o + 10];
    const float* bn2_b   = (const float*)d_in[o + 11];

    thetaphi_kernel<<<M_DIM, 256, 0, stream>>>(X, theta_w, theta_b,
                                               phi_w, phi_b, th, ph);
    graph_softmax_kernel<<<4, 256, 0, stream>>>(th, ph, Cg);
    if (use_tdm) {
      gemm_bf16_wmma<1, true><<<ggrid, 256, 0, stream>>>(
          nullptr, nullptr, Xbf, scW[blk], sconv_b, nullptr, nullptr, S, nullptr);
      mix_bn_relu_kernel<<<(int)(MF / 256), 256, 0, stream>>>(
          S, Cg, bn1_g, bn1_b, H, Hbf);
      gemm_bf16_wmma<2, true><<<ggrid, 256, 0, stream>>>(
          nullptr, nullptr, Hbf, tcW[blk], tconv_b, bn2_g, bn2_b, X, Xbf);
    } else {
      gemm_bf16_wmma<1, false><<<ggrid, 256, 0, stream>>>(
          X, sconv_w, nullptr, nullptr, sconv_b, nullptr, nullptr, S, nullptr);
      mix_bn_relu_kernel<<<(int)(MF / 256), 256, 0, stream>>>(
          S, Cg, bn1_g, bn1_b, H, nullptr);
      gemm_bf16_wmma<2, false><<<ggrid, 256, 0, stream>>>(
          H, tconv_w, nullptr, nullptr, tconv_b, bn2_g, bn2_b, X, nullptr);
    }
  }

  float* out = (float*)d_out;
  action_head_kernel<<<M_DIM, 256, 0, stream>>>(X, act_w, act_b, out);
  activity_head_kernel<<<4 * T_DIM, 256, 0, stream>>>(
      X, acty_w, acty_b, out + 4 * T_DIM * P_DIM * 9);
}
